// RNNTJoiner_49460843380857
// MI455X (gfx1250) — compile-verified, matrix operand order checked by reading, not run on hardware
//
#include <hip/hip_runtime.h>

// ---------------------------------------------------------------------------
// RNN-T joiner: out[b,t,u,v] = relu(src[b,t,:] + tgt[b,u,:]) . W[v,:] + bias[v]
// B=4, T=256, U=64, D=1024, V=1024.  GEMM M=65536, N=1024, K=1024.
// bf16 hi/lo split-precision (3x v_wmma_f32_16x16x32_bf16 per tile, ~fp32 acc).
// W pre-split + pre-swizzled into d_ws once per launch (L2-resident, 4 MB).
// ---------------------------------------------------------------------------

typedef __bf16 bf16;
typedef bf16  bf16x8   __attribute__((ext_vector_type(8)));
typedef bf16  bf16x16  __attribute__((ext_vector_type(16)));
typedef float v8f      __attribute__((ext_vector_type(8)));
typedef float v4f      __attribute__((ext_vector_type(4)));

constexpr int kB = 4, kT = 256, kU = 64, kD = 1024, kV = 1024;
constexpr int kChunks   = kD / 32;     // 32 K-chunks of 32
constexpr int kLdsStride = 72;         // 64 data + 8 pad ushorts (144 B rows)
constexpr size_t kWsNeeded = (size_t)kV * kD * 2 /*bf16*/ * 2 /*hi+lo*/;  // 4 MB

__device__ __forceinline__ v8f wmma_bf16(bf16x16 a, bf16x16 b, v8f c) {
  // (neg_a, A, neg_b, B, c_mod, C, reuse_a, reuse_b)
  return __builtin_amdgcn_wmma_f32_16x16x32_bf16(false, a, false, b,
                                                 (short)0, c, false, false);
}

__device__ __forceinline__ bf16x16 lds_frag(const unsigned short* p) {
  bf16x8 a = *(const bf16x8*)(p);
  bf16x8 b = *(const bf16x8*)(p + 8);
  return __builtin_shufflevector(a, b, 0,1,2,3,4,5,6,7,8,9,10,11,12,13,14,15);
}

// ---------------------------------------------------------------------------
// Pre-pass: split W (V x D f32) into bf16 hi/lo, stored in WMMA B-fragment
// element order per 32-K chunk: [k0..7, k16..23, k8..15, k24..31].
// Lane `half` then reads 16 contiguous bf16 at offset half*16.
// ---------------------------------------------------------------------------
__global__ __launch_bounds__(256)
void wsplit_kernel(const float* __restrict__ W,
                   bf16* __restrict__ whi, bf16* __restrict__ wlo) {
  const int idx = blockIdx.x * 256 + threadIdx.x;   // 0 .. V*kChunks-1
  const int v = idx >> 5;
  const int c = idx & 31;
  const float* wp = W + (size_t)v * kD + c * 32;
  float x[32];
  #pragma unroll
  for (int i = 0; i < 8; ++i) {
    v4f q = *(const v4f*)(wp + i * 4);
    x[i*4+0] = q[0]; x[i*4+1] = q[1]; x[i*4+2] = q[2]; x[i*4+3] = q[3];
  }
  bf16* oh = whi + (size_t)idx * 32;
  bf16* ol = wlo + (size_t)idx * 32;
  const int map[4] = {0, 16, 8, 24};   // fragment-order group -> source k
  #pragma unroll
  for (int g = 0; g < 4; ++g) {
    bf16x8 hv, lv;
    #pragma unroll
    for (int i = 0; i < 8; ++i) {
      float xv = x[map[g] + i];
      bf16 hh = (bf16)xv;
      hv[i] = hh;
      lv[i] = (bf16)(xv - (float)hh);
    }
    *(bf16x8*)(oh + g*8) = hv;
    *(bf16x8*)(ol + g*8) = lv;
  }
}

// ---------------------------------------------------------------------------
// Main GEMM. Workgroup = 8 waves, owns (b, t0..t0+1) -> M-tile 128 rows
// (row = t_local*64 + u) and a 128-wide V slab. Wave owns 16 N-cols, all
// 128 M rows: 8 accumulators. K staged 64 at a time through LDS.
// ---------------------------------------------------------------------------
template <bool USE_WS>
__global__ __launch_bounds__(256)
void rnnt_joiner_wmma(const float* __restrict__ src,   // (B,T,D)
                      const float* __restrict__ tgt,   // (B,U,D)
                      const float* __restrict__ W,     // (V,D)  (fallback)
                      const float* __restrict__ bias,  // (V)
                      const bf16* __restrict__ whi,    // pre-swizzled hi
                      const bf16* __restrict__ wlo,    // pre-swizzled lo
                      float* __restrict__ out)         // (B,T,U,V)
{
  __shared__ __align__(16) unsigned short ldsHi[128 * kLdsStride];
  __shared__ __align__(16) unsigned short ldsLo[128 * kLdsStride];

  const int b    = blockIdx.z;
  const int t0   = blockIdx.y * 2;
  const int v_wg = blockIdx.x * 128;
  const int tid  = threadIdx.x;
  const int wave = tid >> 5;
  const int lane = tid & 31;
  const int l16  = lane & 15;
  const int half = lane >> 4;            // K-half select per WMMA lane layout

  const int v_wave = v_wg + wave * 16;
  const int vcol   = v_wave + l16;

  const float* srcBase = src + (size_t)(b * kT + t0) * kD;   // rows t0, t0+1
  const float* tgtBase = tgt + (size_t)b * kU * kD;
  const float* wRow    = W + (size_t)vcol * kD;              // fallback path

  v8f acc[8];
  #pragma unroll
  for (int i = 0; i < 8; ++i) acc[i] = (v8f){};

  // A staging: each thread produces one row-half: 32 consecutive K values
  const int stRow = tid >> 1;        // 0..127
  const int stT   = stRow >> 6;
  const int stU   = stRow & 63;
  const int stC   = tid & 1;         // which 32-K sub-chunk

  for (int k0 = 0; k0 < kD; k0 += 64) {
    // ---- stage A = relu(src+tgt) as swizzled bf16 hi/lo (128 x 64) ----
    {
      const float* sp = srcBase + (size_t)stT * kD + k0 + stC * 32;
      const float* tp = tgtBase + (size_t)stU * kD + k0 + stC * 32;
      float xs[32];
      #pragma unroll
      for (int i = 0; i < 8; ++i) {
        v4f s = *(const v4f*)(sp + i * 4);
        v4f t = *(const v4f*)(tp + i * 4);
        xs[i*4+0] = s[0] + t[0]; xs[i*4+1] = s[1] + t[1];
        xs[i*4+2] = s[2] + t[2]; xs[i*4+3] = s[3] + t[3];
      }
      unsigned short* oh = &ldsHi[stRow * kLdsStride + stC * 32];
      unsigned short* ol = &ldsLo[stRow * kLdsStride + stC * 32];
      const int map[4] = {0, 16, 8, 24};
      #pragma unroll
      for (int g = 0; g < 4; ++g) {
        bf16x8 hv, lv;
        #pragma unroll
        for (int i = 0; i < 8; ++i) {
          float x = xs[map[g] + i];
          x = x > 0.0f ? x : 0.0f;             // relu
          bf16 hh = (bf16)x;
          hv[i] = hh;
          lv[i] = (bf16)(x - (float)hh);       // residual
        }
        *(bf16x8*)(oh + g*8) = hv;
        *(bf16x8*)(ol + g*8) = lv;
      }
    }
    __syncthreads();

    // ---- compute: 2 sub-chunks x 8 M-tiles x 3 WMMAs ----
    #pragma unroll
    for (int c = 0; c < 2; ++c) {
      bf16x16 wHi, wLo;
      if constexpr (USE_WS) {
        const size_t wo = ((size_t)vcol * kChunks + (k0 >> 5) + c) * 32 + half * 16;
        wHi = lds_frag((const unsigned short*)(whi + wo));  // 2x16B global loads
        wLo = lds_frag((const unsigned short*)(wlo + wo));
      } else {
        const int kb = k0 + c * 32 + half * 8;
        v4f w0 = *(const v4f*)(wRow + kb);
        v4f w1 = *(const v4f*)(wRow + kb + 4);
        v4f w2 = *(const v4f*)(wRow + kb + 16);
        v4f w3 = *(const v4f*)(wRow + kb + 20);
        float wv[16];
        #pragma unroll
        for (int i = 0; i < 4; ++i) {
          wv[i] = w0[i]; wv[4+i] = w1[i]; wv[8+i] = w2[i]; wv[12+i] = w3[i];
        }
        #pragma unroll
        for (int i = 0; i < 16; ++i) {
          bf16 hh = (bf16)wv[i];
          wHi[i] = hh;
          wLo[i] = (bf16)(wv[i] - (float)hh);
        }
      }
      #pragma unroll
      for (int mt = 0; mt < 8; ++mt) {
        const int row = mt * 16 + l16;
        const int ofs = row * kLdsStride + c * 32 + half * 16;
        bf16x16 aHi = lds_frag(&ldsHi[ofs]);
        bf16x16 aLo = lds_frag(&ldsLo[ofs]);
        acc[mt] = wmma_bf16(aHi, wHi, acc[mt]);   // hi*hi
        acc[mt] = wmma_bf16(aHi, wLo, acc[mt]);   // hi*lo
        acc[mt] = wmma_bf16(aLo, wHi, acc[mt]);   // lo*hi
      }
    }
    __syncthreads();
  }

  // ---- epilogue: + bias, coalesced non-temporal stream-out ----
  const float bv = bias[vcol];
  #pragma unroll
  for (int mt = 0; mt < 8; ++mt) {
    #pragma unroll
    for (int i = 0; i < 8; ++i) {
      const int row = mt * 16 + half * 8 + i;   // C/D layout: VGPR i -> M
      const int tt  = t0 + (row >> 6);
      const int uu  = row & 63;
      float* p = out + (((size_t)(b * kT + tt) * kU + uu) * kV + vcol);
      __builtin_nontemporal_store(acc[mt][i] + bv, p);
    }
  }
}

// Tuple tail: reference returns (out, source_lengths, target_lengths)
__global__ void copy_lengths(const int* __restrict__ sl,
                             const int* __restrict__ tl,
                             int* __restrict__ dst) {
  const int i = threadIdx.x;
  if (i < kB)          dst[i] = sl[i];
  else if (i < 2 * kB) dst[i] = tl[i - kB];
}

extern "C" void kernel_launch(void* const* d_in, const int* in_sizes, int n_in,
                              void* d_out, int out_size, void* d_ws, size_t ws_size,
                              hipStream_t stream) {
  const float* src  = (const float*)d_in[0];
  const int*   slen = (const int*)  d_in[1];
  const float* tgt  = (const float*)d_in[2];
  const int*   tlen = (const int*)  d_in[3];
  const float* W    = (const float*)d_in[4];
  const float* bias = (const float*)d_in[5];
  float* out = (float*)d_out;

  dim3 grid(kV / 128, kT / 2, kB);   // (8, 128, 4) = 4096 workgroups

  if (ws_size >= kWsNeeded) {
    bf16* whi = (bf16*)d_ws;
    bf16* wlo = whi + (size_t)kV * kD;
    wsplit_kernel<<<kV * kChunks / 256, 256, 0, stream>>>(W, whi, wlo);
    rnnt_joiner_wmma<true><<<grid, 256, 0, stream>>>(src, tgt, W, bias,
                                                     whi, wlo, out);
  } else {
    rnnt_joiner_wmma<false><<<grid, 256, 0, stream>>>(src, tgt, W, bias,
                                                      nullptr, nullptr, out);
  }

  copy_lengths<<<1, 2 * kB, 0, stream>>>(slen, tlen,
      (int*)out + (size_t)kB * kT * kU * kV);
}
